// CSD_2d_55791625175673
// MI455X (gfx1250) — matrix operand order repaired and checked by reading, not compiled, not run on hardware
//
#include <hip/hip_runtime.h>
#include <hip/hip_bf16.h>

typedef __attribute__((ext_vector_type(16))) _Float16 v16h;
typedef __attribute__((ext_vector_type(8)))  float    v8f;

#define NCLS   4
#define NPIX   131072      /* 8*128*128 */
#define HW     16384
#define DDIM   128
#define NPER   128         /* K/NUM_CLASS, rem = 0 */
#define ROWS_L 516         /* 4*(128+1) */
#define ROWS_LP 528        /* padded to 16 */
#define ROWS_U 1032        /* 4*2*(128+1) */
#define ROWS_UP 1040
#define LDCOL  1040
#define TEMP_INV 10.0f
#define NEG_BIG -1e9f
#define NTH    4096        /* global threads for count/fill */
#define PPT    32          /* pixels per thread */

__device__ __forceinline__ unsigned hash_u32(unsigned x) {
  x ^= x >> 16; x *= 0x7feb352dU; x ^= x >> 15; x *= 0x846ca68bU; x ^= x >> 16;
  return x;
}
__device__ __forceinline__ float u01(unsigned s) {
  return (float)(hash_u32(s) >> 8) * (1.0f / 16777216.0f);
}

/* ---------------- threshold = mean(logits_u) ---------------- */
__global__ void k_thr_part(const float* __restrict__ lg, float* __restrict__ part) {
  __shared__ float s[256];
  int i = blockIdx.x * 256 + threadIdx.x;
  s[threadIdx.x] = lg[i];
  __syncthreads();
  for (int st = 128; st > 0; st >>= 1) {
    if (threadIdx.x < st) s[threadIdx.x] += s[threadIdx.x + st];
    __syncthreads();
  }
  if (!threadIdx.x) part[blockIdx.x] = s[0];
}
__global__ void k_thr_final(const float* __restrict__ part, float* __restrict__ thr) {
  __shared__ float s[512];
  s[threadIdx.x] = part[threadIdx.x];
  __syncthreads();
  for (int st = 256; st > 0; st >>= 1) {
    if (threadIdx.x < st) s[threadIdx.x] += s[threadIdx.x + st];
    __syncthreads();
  }
  if (!threadIdx.x) thr[0] = s[0] / (float)NPIX;
}

/* ---------------- stable compaction: count / scan / fill ---------------- */
__global__ void k_count(const int* __restrict__ pred, const int* __restrict__ labu,
                        const float* __restrict__ logu, const float* __restrict__ thr,
                        int* __restrict__ tcounts) {
  int t = blockIdx.x * 256 + threadIdx.x;      /* 0..4095 */
  int cl[8];
  for (int g = 0; g < 8; ++g) cl[g] = 0;
  float th = thr[0];
  int p0 = t * PPT;
  for (int i = 0; i < PPT; ++i) {
    int p = p0 + i;
    cl[pred[p] & 3]++;
    if (logu[p] >= th) cl[4 + (labu[p] & 3)]++;
  }
  for (int g = 0; g < 8; ++g) tcounts[g * NTH + t] = cl[g];
}

__global__ void k_scan(const int* __restrict__ tcounts, int* __restrict__ toffs,
                       int* __restrict__ gcount) {
  int g = threadIdx.x;
  if (g < 8) {
    int run = 0;
    for (int t = 0; t < NTH; ++t) { toffs[g * NTH + t] = run; run += tcounts[g * NTH + t]; }
    gcount[g] = run;
  }
}

__global__ void k_fill(const int* __restrict__ pred, const int* __restrict__ labu,
                       const float* __restrict__ logu, const float* __restrict__ thr,
                       const int* __restrict__ toffs, const int* __restrict__ gcount,
                       int* __restrict__ listL, int* __restrict__ listU,
                       unsigned char* __restrict__ umap) {
  int t = blockIdx.x * 256 + threadIdx.x;
  int off[8];
  for (int g = 0; g < 8; ++g) off[g] = toffs[g * NTH + t];
  int halfc[4];
  for (int c = 0; c < 4; ++c) halfc[c] = gcount[4 + c] >> 1;
  float th = thr[0];
  int p0 = t * PPT;
  for (int i = 0; i < PPT; ++i) {
    int p = p0 + i;
    int pc = pred[p] & 3;
    listL[pc * NPIX + off[pc]] = p; off[pc]++;
    unsigned char um = 255;
    int lc = labu[p] & 3;
    if (logu[p] >= th) {
      int pos = off[4 + lc]++;
      listU[lc * NPIX + pos] = p;
      um = (pos < halfc[lc]) ? (unsigned char)lc : (unsigned char)(4 + lc);
    }
    umap[p] = um;
  }
}

/* ------------- class-mean features: coalesced full-tensor sweep ------------- */
__global__ void k_mean_part(const float* __restrict__ feat, const int* __restrict__ pred,
                            const unsigned char* __restrict__ umap, int mode,
                            float* __restrict__ part) {
  __shared__ float s[256];
  int e = blockIdx.x * 256 + threadIdx.x;
  int b = e >> 21;                 /* per-image elements = 128*16384 = 2^21 */
  int d = (e >> 14) & 127;
  int rem = e & 16383;
  int n = (b << 14) | rem;
  float v = feat[e];
  int gsel;
  if (mode == 0) gsel = pred[n] & 3;
  else {
    int u = umap[n];
    gsel = (mode == 1) ? ((u < 4) ? u : -1) : ((u >= 4 && u < 8) ? (u - 4) : -1);
  }
  int pidx = (b << 6) | (rem >> 8);           /* 0..511 partial slot per (g,d) */
  for (int g = 0; g < 4; ++g) {
    s[threadIdx.x] = (gsel == g) ? v : 0.f;
    __syncthreads();
    for (int st = 128; st > 0; st >>= 1) {
      if (threadIdx.x < st) s[threadIdx.x] += s[threadIdx.x + st];
      __syncthreads();
    }
    if (!threadIdx.x) part[((g * DDIM + d) << 9) + pidx] = s[0];
    __syncthreads();
  }
}

__global__ void k_mean_reduce(const float* __restrict__ part, const int* __restrict__ gcount,
                              int mode, float* __restrict__ means) {
  __shared__ float s[256];
  int g = blockIdx.x >> 7;
  int d = blockIdx.x & 127;
  int t = threadIdx.x;
  int o = (g * DDIM + d) << 9;
  s[t] = part[o + t] + part[o + t + 256];
  __syncthreads();
  for (int st = 128; st > 0; st >>= 1) {
    if (t < st) s[t] += s[t + st];
    __syncthreads();
  }
  if (!t) {
    int nv;
    if (mode == 0) nv = gcount[g];
    else { int tot = gcount[4 + g]; int h = tot >> 1; nv = (mode == 1) ? h : tot - h; }
    int base = (mode == 0) ? 0 : (mode == 1) ? 4 : 8;
    means[(base + g) * DDIM + d] = s[0] / (float)max(nv, 1);
  }
}

/* ------------- gather sampled rows, convert to f16 ------------- */
__global__ void k_build(const float* __restrict__ fx, const float* __restrict__ fs,
                        const float* __restrict__ ffp,
                        const int* __restrict__ listL, const int* __restrict__ listU,
                        const int* __restrict__ gcount, const float* __restrict__ means,
                        _Float16* __restrict__ featL, _Float16* __restrict__ featU,
                        int* __restrict__ valL, int* __restrict__ valU) {
  int rid = blockIdx.x;
  int d = threadIdx.x;
  if (rid < ROWS_LP) {
    float v = 0.f; int valid = 0;
    if (rid < ROWS_L) {
      int cls = rid / (NPER + 1);
      int j = rid % (NPER + 1);
      int nv = gcount[cls];
      valid = nv > 0;
      if (j == NPER) v = means[cls * DDIM + d];
      else if (nv > 0) {
        int idx;
        if (nv >= NPER) {
          float u = u01(0x13570135u ^ (unsigned)rid);
          idx = (int)(u * (float)nv); if (idx >= nv) idx = nv - 1;
        } else idx = j % nv;
        int p = listL[cls * NPIX + idx];
        v = fx[(((p >> 14) * DDIM + d) << 14) | (p & 16383)];
      }
    }
    featL[rid * DDIM + d] = (_Float16)v;
    if (!d) valL[rid] = valid;
  } else {
    int r = rid - ROWS_LP;
    float v = 0.f; int valid = 0;
    if (r < ROWS_U) {
      int cls = r / (2 * (NPER + 1));
      int within = r % (2 * (NPER + 1));
      int grp = within / (NPER + 1);    /* 0 = s half, 1 = fp half */
      int j = within % (NPER + 1);
      int tot = gcount[4 + cls]; int h = tot >> 1;
      int nv = grp ? (tot - h) : h;
      int base = grp ? h : 0;
      const float* src = grp ? ffp : fs;
      int mbase = grp ? 8 : 4;
      valid = nv > 0;
      if (j == NPER) v = means[(mbase + cls) * DDIM + d];
      else if (nv > 0) {
        int idx;
        if (nv >= NPER) {
          float u = u01(0x2468aceu ^ (unsigned)rid);
          idx = (int)(u * (float)nv); if (idx >= nv) idx = nv - 1;
        } else idx = j % nv;
        int p = listU[cls * NPIX + base + idx];
        v = src[(((p >> 14) * DDIM + d) << 14) | (p & 16383)];
      }
    }
    featU[r * DDIM + d] = (_Float16)v;
    if (!d) valU[r] = valid;
  }
}

/* ------------- GEMM: logits = featL @ featU^T / TEMP via WMMA ------------- */
__global__ __launch_bounds__(32) void k_wmma(const _Float16* __restrict__ A,
                                             const _Float16* __restrict__ B,
                                             float* __restrict__ C) {
  int tm = blockIdx.x;            /* 0..32  -> 528 rows  */
  int tn = blockIdx.y;            /* 0..64  -> 1040 cols */
  int lane = threadIdx.x;
  int r = lane & 15;
  int hl = lane >> 4;
  int arow = tm * 16 + r;
  int brow = tn * 16 + r;         /* C column = row of featU */
  v8f c = {};
#pragma unroll
  for (int kk = 0; kk < DDIM; kk += 32) {
    /* A 16x32 f16 frag: lanes 0-15 K 0-15, lanes 16-31 K 16-31 (contiguous) */
    const _Float16* pa = A + arow * DDIM + kk + hl * 16;
    /* B 32x16 f16 frag: lanes 0-15 K 0-15, lanes 16-31 K 16-31 (contiguous) */
    const _Float16* pb = B + brow * DDIM + kk + hl * 16;
    v16h a, b;
#pragma unroll
    for (int i = 0; i < 16; ++i) a[i] = pa[i];
#pragma unroll
    for (int i = 0; i < 16; ++i) b[i] = pb[i];
    c = __builtin_amdgcn_wmma_f32_16x16x32_f16(false, a, false, b, (short)0, c, false, false);
  }
#pragma unroll
  for (int j = 0; j < 8; ++j) {
    int row = tm * 16 + j + hl * 8;
    int col = tn * 16 + r;
    C[row * LDCOL + col] = c[j] * TEMP_INV;
  }
}

/* ------------- per-anchor masked log-softmax / positive mean ------------- */
__global__ void k_row(const float* __restrict__ logits, const int* __restrict__ valU,
                      float* __restrict__ rowLpp) {
  __shared__ float s[256];
  __shared__ float s2[256];
  __shared__ float s3[256];
  int rrow = blockIdx.x;               /* 0..515 */
  int t = threadIdx.x;
  int clsL = rrow / (NPER + 1);
  const float* lrow = logits + rrow * LDCOL;
  float m = NEG_BIG;
  for (int c = t; c < ROWS_U; c += 256)
    if (valU[c]) m = fmaxf(m, lrow[c]);
  s[t] = m; __syncthreads();
  for (int st = 128; st > 0; st >>= 1) { if (t < st) s[t] = fmaxf(s[t], s[t + st]); __syncthreads(); }
  m = s[0];
  __syncthreads();
  float se = 0.f;
  for (int c = t; c < ROWS_U; c += 256)
    if (valU[c]) se += __expf(lrow[c] - m);
  s[t] = se; __syncthreads();
  for (int st = 128; st > 0; st >>= 1) { if (t < st) s[t] += s[t + st]; __syncthreads(); }
  float ld = __logf(s[0]);
  __syncthreads();
  float ps = 0.f, pc = 0.f;
  for (int c = t; c < ROWS_U; c += 256) {
    if (valU[c] && (c / (2 * (NPER + 1))) == clsL) { ps += lrow[c] - m - ld; pc += 1.f; }
  }
  s2[t] = ps; s3[t] = pc; __syncthreads();
  for (int st = 128; st > 0; st >>= 1) {
    if (t < st) { s2[t] += s2[t + st]; s3[t] += s3[t + st]; }
    __syncthreads();
  }
  if (!t) rowLpp[rrow] = s2[0] / (s3[0] + 1e-12f);
}

__global__ void k_final(const float* __restrict__ rowLpp, const int* __restrict__ valL,
                        const int* __restrict__ valU, float* __restrict__ out) {
  __shared__ float s[512];
  __shared__ int c1[512];
  __shared__ int c2[512];
  int t = threadIdx.x;
  float sum = 0.f; int cnt = 0, any = 0;
  for (int r = t; r < ROWS_L; r += 512) if (valL[r]) { sum += rowLpp[r]; cnt++; }
  for (int c = t; c < ROWS_U; c += 512) any |= valU[c];
  s[t] = sum; c1[t] = cnt; c2[t] = any; __syncthreads();
  for (int st = 256; st > 0; st >>= 1) {
    if (t < st) { s[t] += s[t + st]; c1[t] += c1[t + st]; c2[t] |= c2[t + st]; }
    __syncthreads();
  }
  if (!t) {
    float loss = -s[0] / (float)max(c1[0], 1);
    out[0] = (c1[0] > 0 && c2[0]) ? loss : 0.f;
  }
}

extern "C" void kernel_launch(void* const* d_in, const int* in_sizes, int n_in,
                              void* d_out, int out_size, void* d_ws, size_t ws_size,
                              hipStream_t stream) {
  (void)in_sizes; (void)n_in; (void)out_size; (void)ws_size;
  const float* feat_x    = (const float*)d_in[0];
  const int*   pred_gt   = (const int*)d_in[1];
  /* d_in[2] = gt : unused by the reference */
  const float* feat_u_s  = (const float*)d_in[3];
  const float* feat_u_fp = (const float*)d_in[4];
  const float* logits_u  = (const float*)d_in[5];
  const int*   label_u   = (const int*)d_in[6];

  char* wsp = (char*)d_ws;
  size_t off = 0;
  auto take = [&](size_t bytes) -> void* {
    void* p = wsp + off;
    off += (bytes + 255) & ~(size_t)255;
    return p;
  };
  float* thrPart = (float*)take(512 * sizeof(float));
  float* thrVal  = (float*)take(sizeof(float));
  int* tcounts   = (int*)take(8 * NTH * sizeof(int));
  int* toffs     = (int*)take(8 * NTH * sizeof(int));
  int* gcount    = (int*)take(16 * sizeof(int));
  int* listL     = (int*)take(4 * (size_t)NPIX * sizeof(int));
  int* listU     = (int*)take(4 * (size_t)NPIX * sizeof(int));
  unsigned char* umap = (unsigned char*)take(NPIX);
  float* part    = (float*)take(4 * 128 * 512 * sizeof(float));
  float* means   = (float*)take(12 * 128 * sizeof(float));
  _Float16* featL = (_Float16*)take((size_t)ROWS_LP * DDIM * sizeof(_Float16));
  _Float16* featU = (_Float16*)take((size_t)ROWS_UP * DDIM * sizeof(_Float16));
  int* valL      = (int*)take(ROWS_LP * sizeof(int));
  int* valU      = (int*)take(ROWS_UP * sizeof(int));
  float* logitsB = (float*)take((size_t)ROWS_LP * LDCOL * sizeof(float));
  float* rowLpp  = (float*)take(ROWS_LP * sizeof(float));

  k_thr_part <<<512, 256, 0, stream>>>(logits_u, thrPart);
  k_thr_final<<<1, 512, 0, stream>>>(thrPart, thrVal);
  k_count    <<<16, 256, 0, stream>>>(pred_gt, label_u, logits_u, thrVal, tcounts);
  k_scan     <<<1, 32, 0, stream>>>(tcounts, toffs, gcount);
  k_fill     <<<16, 256, 0, stream>>>(pred_gt, label_u, logits_u, thrVal, toffs, gcount,
                                      listL, listU, umap);
  /* three coalesced mean sweeps (labeled / unlab-s / unlab-fp), part buffer reused */
  k_mean_part  <<<65536, 256, 0, stream>>>(feat_x,    pred_gt, umap, 0, part);
  k_mean_reduce<<<512, 256, 0, stream>>>(part, gcount, 0, means);
  k_mean_part  <<<65536, 256, 0, stream>>>(feat_u_s,  pred_gt, umap, 1, part);
  k_mean_reduce<<<512, 256, 0, stream>>>(part, gcount, 1, means);
  k_mean_part  <<<65536, 256, 0, stream>>>(feat_u_fp, pred_gt, umap, 2, part);
  k_mean_reduce<<<512, 256, 0, stream>>>(part, gcount, 2, means);

  k_build<<<ROWS_LP + ROWS_UP, 128, 0, stream>>>(feat_x, feat_u_s, feat_u_fp,
                                                 listL, listU, gcount, means,
                                                 featL, featU, valL, valU);
  k_wmma <<<dim3(ROWS_LP / 16, ROWS_UP / 16), 32, 0, stream>>>(featL, featU, logitsB);
  k_row  <<<ROWS_L, 256, 0, stream>>>(logitsB, valU, rowLpp);
  k_final<<<1, 512, 0, stream>>>(rowLpp, valL, valU, (float*)d_out);
}
